// DeformableConv2d_14310831030801
// MI455X (gfx1250) — matrix operand order repaired and checked by reading, not compiled, not run on hardware
//
#include <hip/hip_runtime.h>
#include <hip/hip_bf16.h>
#include <math.h>

// ---------------------------------------------------------------------------
// CDNA5 (gfx1250) implementation of the deformable-conv UNet reference.
// All 3x3 convolutions run as implicit-GEMM on V_WMMA_F32_16X16X32_BF16:
//   - 256-thread blocks = 8 wave32s; 16(Co) x 128(pixels) tile per block
//   - one shared 16x32 fp32 weight tile per K-chunk, staged by the Tensor
//     Data Mover (TENSOR_LOAD_TO_LDS + s_wait_tensorcnt) when available
//   - each wave stages its own 32x16 bf16 im2col subtile and issues WMMA
// Gather-heavy stages (bilinear deform sampling, 1x1 head, pool/upsample/
// concat, instance-norm) are bandwidth-bound scalar kernels.
// ---------------------------------------------------------------------------

typedef __bf16 bf16x16 __attribute__((ext_vector_type(16)));
typedef float  f32x8   __attribute__((ext_vector_type(8)));
typedef unsigned int u32x4 __attribute__((ext_vector_type(4)));
typedef int          i32x4 __attribute__((ext_vector_type(4)));
typedef int          i32x8 __attribute__((ext_vector_type(8)));

#if defined(__HIP_DEVICE_COMPILE__) && __has_builtin(__builtin_amdgcn_tensor_load_to_lds)
#define HAVE_TDM 1
#else
#define HAVE_TDM 0
#endif

static __device__ __forceinline__ unsigned short f32_to_bf16_rne(float f) {
  union { float f; unsigned int u; } v; v.f = f;
  unsigned int r = v.u + 0x7FFFu + ((v.u >> 16) & 1u);  // round-to-nearest-even
  return (unsigned short)(r >> 16);
}

// ---------------------------------------------------------------------------
// Implicit-GEMM 3x3 conv (pad=1, stride=1, NCHW, no bias).
//   A (16xK)  = weights  [Co tile x (Ci*9)]   (K index enumerates (ci,kh,kw),
//               which is exactly the contiguous OIHW row: row stride = Ci*9)
//   B (Kx16)  = im2col   [(Ci*9) x pixel tile]
//   D (16x16) = output   [Co tile x pixel tile], fp32 accum
// grid = (HW/128, Co/16, B); block = 256 (8 waves, one 16-pixel subtile each).
// CinStride/cBase allow convolving a channel-slice of a larger NCHW tensor
// (used for line_fea = x[:, 3:6]).
// ---------------------------------------------------------------------------
__global__ __launch_bounds__(256) void conv3x3_wmma_k(
    const float* __restrict__ X, const float* __restrict__ Wt,
    float* __restrict__ Y,
    int Ci, int CinStride, int cBase, int Co, int H, int W)
{
  const int tid  = threadIdx.x;
  const int lane = tid & 31;
  const int wave = tid >> 5;                     // 0..7
  const int n0   = blockIdx.x * 128 + wave * 16; // this wave's 16-pixel subtile
  const int co0  = blockIdx.y * 16;
  const int b    = blockIdx.z;
  const int HW   = H * W;
  const int Ktot = Ci * 9;

  __shared__ float          Asf[16 * 32];      // shared A chunk, fp32 (TDM dest)
  __shared__ unsigned short Bs[8][32 * 16];    // per-wave B chunk, bf16 bits

  f32x8 acc = {0.f, 0.f, 0.f, 0.f, 0.f, 0.f, 0.f, 0.f};

  for (int k0 = 0; k0 < Ktot; k0 += 32) {
    // --- stage A: 16 x 32 fp32 weight tile --------------------------------
#if HAVE_TDM
    if (wave == 0) {
      // 2D Tensor-DMA descriptor (CDNA5 ISA):
      //   tile 32(K) x 16(rows), tensor_dim0 = Ktot-k0 (OOB reads -> 0 pads
      //   the K tail), dim0 stride = Ktot, data_size = 4B, type = 2 (image).
      unsigned long long ga =
          (unsigned long long)(const void*)(Wt + (size_t)co0 * Ktot + k0);
      unsigned lds = (unsigned)(unsigned long long)(const void*)&Asf[0];
      unsigned rem = (unsigned)(Ktot - k0);
      u32x4 g0;
      g0[0] = 1u;                                        // count=1, user mode
      g0[1] = lds;                                       // lds_addr
      g0[2] = (unsigned)(ga & 0xffffffffu);              // global_addr[31:0]
      g0[3] = (unsigned)((ga >> 32) & 0x01ffffffu)       // global_addr[56:32]
              | 0x80000000u;                             // type=2 ("image")
      i32x8 g1;
      g1[0] = (int)(2u << 16);                           // data_size=4B; no mask
      g1[1] = (int)((rem & 0xffffu) << 16);              // tensor_dim0[15:0]
      g1[2] = (int)((rem >> 16) | (16u << 16));          // td0[31:16] | tensor_dim1=16
      g1[3] = (int)(32u << 16);                          // td1[31:16]=0 | tile_dim0=32
      g1[4] = 16;                                        // tile_dim1=16, tile_dim2=0
      g1[5] = (int)(unsigned)Ktot;                       // dim0_stride[31:0]
      g1[6] = 0;                                         // dim0_stride[47:32], dim1_stride lo
      g1[7] = 0;
      i32x4 g2 = {0, 0, 0, 0};                           // tensor_dim2/3 unused (2D)
      i32x4 g3 = {0, 0, 0, 0};
#if __clang_major__ >= 23
      i32x8 g4 = {0, 0, 0, 0, 0, 0, 0, 0};
      __builtin_amdgcn_tensor_load_to_lds(g0, g1, g2, g3, g4, 0);
#else
      __builtin_amdgcn_tensor_load_to_lds(g0, g1, g2, g3, 0);
#endif
    }
#else
    for (int i = tid; i < 512; i += 256) {
      int m = i >> 5;
      int k = (i & 31) + k0;
      Asf[i] = (k < Ktot) ? Wt[(size_t)(co0 + m) * Ktot + k] : 0.f;
    }
#endif

    // --- stage B: per-wave 32 x 16 bf16 im2col tile (zero halo / K tail) ---
    for (int i = lane; i < 512; i += 32) {
      int k = (i >> 4) + k0;
      int n = i & 15;
      float v = 0.f;
      if (k < Ktot) {
        int ci = k / 9, r = k - ci * 9;
        int kh = r / 3, kw = r - kh * 3;
        int pix = n0 + n;
        int oy = pix / W, ox = pix - oy * W;
        int iy = oy - 1 + kh, ix = ox - 1 + kw;
        if (iy >= 0 && iy < H && ix >= 0 && ix < W)
          v = X[((size_t)(b * CinStride + cBase + ci) * H + iy) * W + ix];
      }
      Bs[wave][i] = f32_to_bf16_rne(v);
    }

#if HAVE_TDM
    if (wave == 0) __builtin_amdgcn_s_wait_tensorcnt(0);  // A tile landed in LDS
#endif
    __syncthreads();

    // --- gather fragments per the CDNA5 16-bit WMMA VGPR layout -------------
    // A 16x32: lane<16 holds row=lane, K {0..7}U{16..23}; lane>=16 holds
    //          row=lane-16, K {8..15}U{24..31}; 2 values per VGPR.
    // B 32x16: lane<16 holds col=lane, K {0..7}U{16..23}; lane>=16 col=lane-16, K {8..15}U{24..31}.
    union { bf16x16 v; unsigned short u[16]; } fa, fb;
    const int half = lane >> 4;
    const int rc   = lane & 15;       // A row / B column
    #pragma unroll
    for (int g = 0; g < 8; ++g) {
      int kk = ((g < 4) ? (2 * g) : (16 + 2 * (g - 4))) + 8 * half;
      fa.u[2 * g]     = f32_to_bf16_rne(Asf[rc * 32 + kk]);
      fa.u[2 * g + 1] = f32_to_bf16_rne(Asf[rc * 32 + kk + 1]);
    }
    const int kb = 8 * half;
    #pragma unroll
    for (int g = 0; g < 8; ++g) {
      int kk = ((g < 4) ? (2 * g) : (16 + 2 * (g - 4))) + kb;
      fb.u[2 * g]     = Bs[wave][(kk) * 16 + rc];
      fb.u[2 * g + 1] = Bs[wave][(kk + 1) * 16 + rc];
    }

#if defined(__HIP_DEVICE_COMPILE__)
    acc = __builtin_amdgcn_wmma_f32_16x16x32_bf16(
        /*neg_a=*/false, fa.v, /*neg_b=*/false, fb.v,
        /*c_mod=*/(short)0, acc, /*reuse_a=*/false, /*reuse_b=*/false);
#endif
    __syncthreads();
  }

  // D layout: VGPR r -> M = r + 8*(lane>=16), N = lane&15.
  const int pix = n0 + (lane & 15);
  const int mb  = 8 * (lane >> 4);
  #pragma unroll
  for (int r = 0; r < 8; ++r)
    Y[((size_t)(b * Co + co0 + mb + r)) * HW + pix] = acc[r];
}

// ---------------------------------------------------------------------------
// In-place instance-norm + ReLU over one (b, c) plane per block.
// ---------------------------------------------------------------------------
__global__ __launch_bounds__(256) void inorm_relu_k(float* __restrict__ X, int HW)
{
  float* p = X + (size_t)blockIdx.x * HW;
  __shared__ float rs[256], rq[256];
  float s = 0.f, q = 0.f;
  for (int i = threadIdx.x; i < HW; i += 256) { float v = p[i]; s += v; q += v * v; }
  rs[threadIdx.x] = s; rq[threadIdx.x] = q;
  __syncthreads();
  for (int o = 128; o > 0; o >>= 1) {
    if (threadIdx.x < o) { rs[threadIdx.x] += rs[threadIdx.x + o]; rq[threadIdx.x] += rq[threadIdx.x + o]; }
    __syncthreads();
  }
  const float mean = rs[0] / (float)HW;
  const float var  = rq[0] / (float)HW - mean * mean;
  const float inv  = rsqrtf(var + 1e-5f);
  for (int i = threadIdx.x; i < HW; i += 256) {
    float v = (p[i] - mean) * inv;
    p[i] = v > 0.f ? v : 0.f;
  }
}

// ---------------------------------------------------------------------------
__global__ __launch_bounds__(256) void maxpool2_k(
    const float* __restrict__ X, float* __restrict__ O, int BC, int Hi, int Wi)
{
  const int Ho = Hi >> 1, Wo = Wi >> 1;
  size_t i = (size_t)blockIdx.x * 256 + threadIdx.x;
  size_t total = (size_t)BC * Ho * Wo;
  if (i >= total) return;
  int xo = (int)(i % Wo); size_t t = i / Wo;
  int yo = (int)(t % Ho); int bc = (int)(t / Ho);
  const float* p = X + ((size_t)bc * Hi + 2 * yo) * Wi + 2 * xo;
  O[i] = fmaxf(fmaxf(p[0], p[1]), fmaxf(p[Wi], p[Wi + 1]));
}

__global__ __launch_bounds__(256) void upsample2_k(
    const float* __restrict__ X, float* __restrict__ O, int BC, int Ho, int Wo)
{
  size_t i = (size_t)blockIdx.x * 256 + threadIdx.x;
  size_t total = (size_t)BC * Ho * Wo;
  if (i >= total) return;
  int xo = (int)(i % Wo); size_t t = i / Wo;
  int yo = (int)(t % Ho); int bc = (int)(t / Ho);
  O[i] = X[((size_t)bc * (Ho >> 1) + (yo >> 1)) * (Wo >> 1) + (xo >> 1)];
}

__global__ __launch_bounds__(256) void concat2_k(
    const float* __restrict__ A, const float* __restrict__ Bp, float* __restrict__ O,
    int Bn, int C1, int C2, int HW)
{
  size_t i = (size_t)blockIdx.x * 256 + threadIdx.x;
  size_t total = (size_t)Bn * (C1 + C2) * HW;
  if (i >= total) return;
  int hw = (int)(i % HW); size_t t = i / HW;
  int c = (int)(t % (C1 + C2)); int b = (int)(t / (C1 + C2));
  O[i] = (c < C1) ? A[((size_t)b * C1 + c) * HW + hw]
                  : Bp[((size_t)b * C2 + (c - C1)) * HW + hw];
}

// 1x1 conv (16 -> 18) with bias and offset clip to [-clipv, clipv].
__global__ __launch_bounds__(256) void conv1x1_clip_k(
    const float* __restrict__ X, const float* __restrict__ Wt,
    const float* __restrict__ Bs, float* __restrict__ O,
    int Bn, int Ci, int Co, int HW, float clipv)
{
  size_t idx = (size_t)blockIdx.x * 256 + threadIdx.x;
  if (idx >= (size_t)Bn * HW) return;
  int b = (int)(idx / HW), pix = (int)(idx % HW);
  float xin[16];
  for (int ci = 0; ci < Ci; ++ci) xin[ci] = X[((size_t)(b * Ci + ci)) * HW + pix];
  for (int o = 0; o < Co; ++o) {
    float a = Bs[o];
    for (int ci = 0; ci < Ci; ++ci) a += Wt[o * Ci + ci] * xin[ci];
    a = fminf(fmaxf(a, -clipv), clipv);
    O[((size_t)(b * Co + o)) * HW + pix] = a;
  }
}

// Deformable 3x3 conv: 3 input channels (slice of 6-ch x), 18-ch offsets,
// 32 output channels. One thread per (b, pixel): 27 bilinear samples cached
// in registers, then 32x27 FMA fan-out.
__global__ __launch_bounds__(256) void deform_conv_k(
    const float* __restrict__ X, int Ctot, int cBase,
    const float* __restrict__ OFF, const float* __restrict__ Wc,
    float* __restrict__ O, int Bn, int H, int W)
{
  const int HW = H * W;
  size_t idx = (size_t)blockIdx.x * 256 + threadIdx.x;
  if (idx >= (size_t)Bn * HW) return;
  const int b = (int)(idx / HW), pix = (int)(idx % HW);
  const int oy = pix / W, ox = pix - oy * W;

  float samp[27];
  #pragma unroll
  for (int k = 0; k < 9; ++k) {
    const float dy = OFF[((size_t)(b * 18 + 2 * k)) * HW + pix];
    const float dx = OFF[((size_t)(b * 18 + 2 * k + 1)) * HW + pix];
    const float ys = (float)(oy - 1 + k / 3) + dy;
    const float xs = (float)(ox - 1 + k % 3) + dx;
    const float y0f = floorf(ys), x0f = floorf(xs);
    const float wy = ys - y0f, wx = xs - x0f;
    const int y0 = (int)y0f, x0 = (int)x0f;
    const float w00 = (1.f - wy) * (1.f - wx), w01 = (1.f - wy) * wx;
    const float w10 = wy * (1.f - wx),        w11 = wy * wx;
    const bool vy0 = (y0 >= 0) & (y0 < H),      vy1 = (y0 + 1 >= 0) & (y0 + 1 < H);
    const bool vx0 = (x0 >= 0) & (x0 < W),      vx1 = (x0 + 1 >= 0) & (x0 + 1 < W);
    const int y0c = min(max(y0, 0), H - 1),     y1c = min(max(y0 + 1, 0), H - 1);
    const int x0c = min(max(x0, 0), W - 1),     x1c = min(max(x0 + 1, 0), W - 1);
    #pragma unroll
    for (int c = 0; c < 3; ++c) {
      const float* pc = X + ((size_t)(b * Ctot + cBase + c)) * HW;
      float v = 0.f;
      if (vy0 & vx0) v += w00 * pc[y0c * W + x0c];
      if (vy0 & vx1) v += w01 * pc[y0c * W + x1c];
      if (vy1 & vx0) v += w10 * pc[y1c * W + x0c];
      if (vy1 & vx1) v += w11 * pc[y1c * W + x1c];
      samp[c * 9 + k] = v;
    }
  }
  #pragma unroll 4
  for (int o = 0; o < 32; ++o) {
    float acc = 0.f;
    #pragma unroll
    for (int c = 0; c < 3; ++c)
      #pragma unroll
      for (int k = 0; k < 9; ++k)
        acc += samp[c * 9 + k] * Wc[(o * 3 + c) * 9 + k];
    O[((size_t)(b * 32 + o)) * HW + pix] = acc;
  }
}

// ---------------------------------------------------------------------------
extern "C" void kernel_launch(void* const* d_in, const int* in_sizes, int n_in,
                              void* d_out, int out_size, void* d_ws, size_t ws_size,
                              hipStream_t stream)
{
  (void)in_sizes; (void)n_in; (void)out_size; (void)ws_size;
  const int B = 4, H0 = 256, W0 = 256;
  const float* x = (const float*)d_in[0];
  auto P = [&](int i) { return (const float*)d_in[i]; };
  // d_in order (setup_inputs dict order): x, then params by insertion:
  // 1 c1_w1 2 c1_b1 3 c1_w2 4 c1_b2 | 5..8 c2 | 9..12 c3 | 13..16 c4
  // 17 u4_w 18 u4_b | 19..22 uc4 | 23 u3_w 24 u3_b | 25..28 uc3
  // 29 u2_w 30 u2_b | 31..34 uc2 | 35 c1x1_w 36 c1x1_b
  // 37 line_w 38 color_w 39 out_w 40 out_b

  // bump allocator over d_ws
  size_t off_b = 0;
  auto alloc = [&](size_t elems) -> float* {
    float* p = (float*)((char*)d_ws + off_b);
    off_b += (elems * sizeof(float) + 255) & ~(size_t)255;
    return p;
  };

  auto conv3 = [&](const float* in, const float* wt, float* out,
                   int Ci, int Cstride, int cBase, int Co, int h, int w) {
    dim3 g((h * w) / 128, Co / 16, B);
    conv3x3_wmma_k<<<g, dim3(256), 0, stream>>>(in, wt, out, Ci, Cstride, cBase, Co, h, w);
  };
  auto inorm = [&](float* buf, int C, int hw) {
    inorm_relu_k<<<dim3(B * C), dim3(256), 0, stream>>>(buf, hw);
  };
  auto cb = [&](const float* in, int Ci, int Co, int h, int w,
                const float* w1, const float* w2, float* t1, float* outb) {
    conv3(in, w1, t1, Ci, Ci, 0, Co, h, w);  inorm(t1, Co, h * w);
    conv3(t1, w2, outb, Co, Co, 0, Co, h, w); inorm(outb, Co, h * w);
  };
  auto pool = [&](const float* in, float* outb, int C, int hi, int wi) {
    size_t tot = (size_t)B * C * (hi / 2) * (wi / 2);
    maxpool2_k<<<dim3((unsigned)((tot + 255) / 256)), dim3(256), 0, stream>>>(in, outb, B * C, hi, wi);
  };
  auto up2 = [&](const float* in, float* outb, int C, int ho, int wo) {
    size_t tot = (size_t)B * C * ho * wo;
    upsample2_k<<<dim3((unsigned)((tot + 255) / 256)), dim3(256), 0, stream>>>(in, outb, B * C, ho, wo);
  };
  auto cat = [&](const float* a, const float* bp, float* outb, int C1, int C2, int hw) {
    size_t tot = (size_t)B * (C1 + C2) * hw;
    concat2_k<<<dim3((unsigned)((tot + 255) / 256)), dim3(256), 0, stream>>>(a, bp, outb, B, C1, C2, hw);
  };

  const int HW = H0 * W0;           // 65536
  float* tmp  = alloc((size_t)B * 16 * HW);       // largest cb temp (16ch @ 256^2)
  float* x1   = alloc((size_t)B * 16 * HW);
  float* p1   = alloc((size_t)B * 16 * HW / 4);
  float* x2   = alloc((size_t)B * 32 * HW / 4);
  float* p2   = alloc((size_t)B * 32 * HW / 16);
  float* x3   = alloc((size_t)B * 64 * HW / 16);
  float* p3   = alloc((size_t)B * 64 * HW / 64);
  float* x4   = alloc((size_t)B * 128 * HW / 64);
  float* up4  = alloc((size_t)B * 128 * HW / 16);
  float* u4o  = alloc((size_t)B * 64 * HW / 16);
  float* cat4 = alloc((size_t)B * 128 * HW / 16);
  float* d4   = alloc((size_t)B * 64 * HW / 16);
  float* up3  = alloc((size_t)B * 64 * HW / 4);
  float* u3o  = alloc((size_t)B * 32 * HW / 4);
  float* cat3 = alloc((size_t)B * 64 * HW / 4);
  float* d3   = alloc((size_t)B * 32 * HW / 4);
  float* upb2 = alloc((size_t)B * 32 * HW);
  float* u2o  = alloc((size_t)B * 16 * HW);
  float* cat2 = alloc((size_t)B * 32 * HW);
  float* d2   = alloc((size_t)B * 16 * HW);
  float* offs = alloc((size_t)B * 18 * HW);
  float* colr = alloc((size_t)B * 32 * HW);
  float* line = alloc((size_t)B * 32 * HW);
  float* catO = alloc((size_t)B * 64 * HW);

  // ---- UNet encoder ----
  cb(x, 6, 16, 256, 256, P(1), P(3), tmp, x1);
  pool(x1, p1, 16, 256, 256);
  cb(p1, 16, 32, 128, 128, P(5), P(7), tmp, x2);
  pool(x2, p2, 32, 128, 128);
  cb(p2, 32, 64, 64, 64, P(9), P(11), tmp, x3);
  pool(x3, p3, 64, 64, 64);
  cb(p3, 64, 128, 32, 32, P(13), P(15), tmp, x4);

  // ---- UNet decoder ----
  up2(x4, up4, 128, 64, 64);
  conv3(up4, P(17), u4o, 128, 128, 0, 64, 64, 64); inorm(u4o, 64, 64 * 64);
  cat(x3, u4o, cat4, 64, 64, 64 * 64);
  cb(cat4, 128, 64, 64, 64, P(19), P(21), tmp, d4);

  up2(d4, up3, 64, 128, 128);
  conv3(up3, P(23), u3o, 64, 64, 0, 32, 128, 128); inorm(u3o, 32, 128 * 128);
  cat(x2, u3o, cat3, 32, 32, 128 * 128);
  cb(cat3, 64, 32, 128, 128, P(25), P(27), tmp, d3);

  up2(d3, upb2, 32, 256, 256);
  conv3(upb2, P(29), u2o, 32, 32, 0, 16, 256, 256); inorm(u2o, 16, HW);
  cat(x1, u2o, cat2, 16, 16, HW);
  cb(cat2, 32, 16, 256, 256, P(31), P(33), tmp, d2);

  // ---- offset head: 1x1 conv + clip(+-max(H,W)/4 = 64) ----
  {
    size_t tot = (size_t)B * HW;
    conv1x1_clip_k<<<dim3((unsigned)((tot + 255) / 256)), dim3(256), 0, stream>>>(
        d2, P(35), P(36), offs, B, 16, 18, HW, 64.0f);
  }

  // ---- deformable conv on color channels x[:, :3], plain conv on x[:, 3:6] ----
  {
    size_t tot = (size_t)B * HW;
    deform_conv_k<<<dim3((unsigned)((tot + 255) / 256)), dim3(256), 0, stream>>>(
        x, 6, 0, offs, P(38), colr, B, H0, W0);
  }
  conv3(x, P(37), line, 3, 6, 3, 32, 256, 256);   // line_w, no bias

  // ---- fuse: concat -> inorm/relu -> conv out_w -> inorm/relu -> d_out ----
  cat(colr, line, catO, 32, 32, HW);
  inorm(catO, 64, HW);
  conv3(catO, P(39), (float*)d_out, 64, 64, 0, 64, 256, 256);
  inorm((float*)d_out, 64, HW);
}
